// PhaseMultiHeadDecoder_14121852469307
// MI455X (gfx1250) — matrix operand order repaired and checked by reading, not compiled, not run on hardware
//
#include <hip/hip_runtime.h>

#define B_  4
#define S_  1024
#define E_  1024
#define H_  16
#define HD_ 64
#define V_  256
#define EPS_ 1e-5f

typedef __bf16 bf16_t;
typedef __attribute__((ext_vector_type(8)))  __bf16 v8bf;
typedef __attribute__((ext_vector_type(16))) __bf16 v16bf;
typedef __attribute__((ext_vector_type(8)))  float  v8f;
typedef __attribute__((ext_vector_type(4)))  unsigned int v4u;
typedef __attribute__((ext_vector_type(4)))  int v4i;
typedef __attribute__((ext_vector_type(8)))  int v8i;

#if defined(__has_builtin)
#if __has_builtin(__builtin_amdgcn_tensor_load_to_lds) && \
    __has_builtin(__builtin_amdgcn_s_wait_tensorcnt)
#define HAVE_TDM 1
#endif
#endif
#ifndef HAVE_TDM
#define HAVE_TDM 0
#endif

static __device__ __forceinline__ v8f vzero8() {
  v8f z;
#pragma unroll
  for (int i = 0; i < 8; ++i) z[i] = 0.0f;
  return z;
}

// A-operand fragment (16x32 bf16): lane holds one row; K chunks
// {0..7,16..23} for lanes 0-15, {8..15,24..31} for lanes 16-31 (ISA 7.12.2).
static __device__ __forceinline__ v16bf frag_a(const bf16_t* rowk0, int lane) {
  const bf16_t* p = rowk0 + ((lane & 16) ? 8 : 0);
  v8bf lo = *(const v8bf*)p;
  v8bf hi = *(const v8bf*)(p + 16);
  return __builtin_shufflevector(lo, hi, 0, 1, 2, 3, 4, 5, 6, 7,
                                 8, 9, 10, 11, 12, 13, 14, 15);
}

// B-operand fragment (32x16 bf16, source stored as rows of B^T = [N][K]):
// lane holds column n=lane%16; K chunks {0..7,16..23} (lanes 0-15) /
// {8..15,24..31} (lanes 16-31), matching the A-fragment K order.
static __device__ __forceinline__ v16bf frag_b(const bf16_t* rowk0, int lane) {
  const bf16_t* p = rowk0 + ((lane & 16) ? 8 : 0);
  v8bf lo = *(const v8bf*)p;
  v8bf hi = *(const v8bf*)(p + 16);
  return __builtin_shufflevector(lo, hi, 0, 1, 2, 3, 4, 5, 6, 7,
                                 8, 9, 10, 11, 12, 13, 14, 15);
}

static __device__ __forceinline__ v8f wmma_bf16(v16bf a, v16bf b, v8f c) {
  return __builtin_amdgcn_wmma_f32_16x16x32_bf16(false, a, false, b,
                                                 (short)0, c, false, false);
}

#if HAVE_TDM
// Issue one TDM 2-D tile load: `rows` rows of `rowlen` bf16 elements from a
// row-major tensor with row stride `stride` elements, into contiguous LDS.
// D# layout per ISA 8.3/8.4 (count=1, type=2, data_size=2B).
static __device__ __forceinline__ void tdm_load_tile(unsigned lds_off,
                                                     const bf16_t* gaddr,
                                                     int rows, int rowlen,
                                                     int stride) {
  unsigned long long ga = (unsigned long long)(uintptr_t)gaddr;
  v4u g0;
  g0[0] = 1u;  // count=1, is_restore=0, gather=0
  g0[1] = lds_off;
  g0[2] = (unsigned)ga;
  g0[3] = (unsigned)((ga >> 32) & 0x01FFFFFFull) | 0x80000000u;  // type=2
  v8i g1;
  const unsigned td0 = (unsigned)E_;  // tensor_dim0 (elements)
  const unsigned td1 = (unsigned)E_;  // tensor_dim1 (rows)
  g1[0] = 0x00010000;                            // data_size=1 -> 2 bytes
  g1[1] = (int)((td0 & 0xFFFFu) << 16);          // tensor_dim0[15:0]
  g1[2] = (int)(((td0 >> 16) & 0xFFFFu) | ((td1 & 0xFFFFu) << 16));
  g1[3] = (int)(((td1 >> 16) & 0xFFFFu) |
                (((unsigned)rowlen & 0xFFFFu) << 16));  // tile_dim0
  g1[4] = rows & 0xFFFF;                         // tile_dim1 (tile_dim2=0)
  g1[5] = stride;                                // tensor_dim0_stride[31:0]
  g1[6] = 0;
  g1[7] = 0;
  v4i z4;
  z4[0] = 0; z4[1] = 0; z4[2] = 0; z4[3] = 0;
#if __clang_major__ >= 23
  v8i z8;
#pragma unroll
  for (int i = 0; i < 8; ++i) z8[i] = 0;
  __builtin_amdgcn_tensor_load_to_lds(g0, g1, z4, z4, z8, 0);
#else
  __builtin_amdgcn_tensor_load_to_lds(g0, g1, z4, z4, 0);
#endif
}
#endif

// ---------------------------------------------------------------------------
// Kernel 1: embedding + positional phase + Q/K/V rotations.
// ---------------------------------------------------------------------------
__global__ __launch_bounds__(256) void embed_kernel(
    const int* __restrict__ x, const float* __restrict__ emb,
    const float* __restrict__ posp,
    const float* __restrict__ qrot, const float* __restrict__ krot,
    const float* __restrict__ vrot,
    float* __restrict__ sr, float* __restrict__ si,
    bf16_t* __restrict__ qr, bf16_t* __restrict__ qi,
    bf16_t* __restrict__ kr, bf16_t* __restrict__ ki,
    bf16_t* __restrict__ vtr, bf16_t* __restrict__ vti) {
  size_t idx = (size_t)blockIdx.x * 256 + threadIdx.x;  // over B*S*E
  int e = (int)(idx & (E_ - 1));
  int s = (int)((idx >> 10) & (S_ - 1));
  int b = (int)(idx >> 20);
  int tok = x[b * S_ + s];
  float mag = tanhf(emb[(size_t)tok * E_ + e]);
  float ph = posp[(size_t)s * E_ + e];
  float sn, cs;
  __sincosf(ph, &sn, &cs);
  float r0 = mag * cs, i0 = mag * sn;
  sr[idx] = r0;
  si[idx] = i0;
  float sa, ca;
  __sincosf(qrot[e], &sa, &ca);  // q_rot[h,d] flattened == index e
  qr[idx] = (bf16_t)(r0 * ca - i0 * sa);
  qi[idx] = (bf16_t)(r0 * sa + i0 * ca);
  __sincosf(krot[e], &sa, &ca);
  kr[idx] = (bf16_t)(r0 * ca - i0 * sa);
  ki[idx] = (bf16_t)(r0 * sa + i0 * ca);
  __sincosf(vrot[e], &sa, &ca);
  int h = e >> 6, d = e & 63;
  size_t tv = ((size_t)((b * H_ + h) * HD_ + d)) * S_ + s;
  vtr[tv] = (bf16_t)(r0 * ca - i0 * sa);
  vti[tv] = (bf16_t)(r0 * sa + i0 * ca);
}

// ---------------------------------------------------------------------------
// Kernel 2: transpose-convert ff -> bf16 ff^T; convert readout weights.
// ---------------------------------------------------------------------------
__global__ __launch_bounds__(256) void prep_kernel(
    const float* __restrict__ ffr, const float* __restrict__ ffi,
    const float* __restrict__ rwr, const float* __restrict__ rwi,
    bf16_t* __restrict__ fTr, bf16_t* __restrict__ fTi,
    bf16_t* __restrict__ wrb, bf16_t* __restrict__ wib) {
  size_t idx = (size_t)blockIdx.x * 256 + threadIdx.x;  // over E*E
  size_t e = idx & (E_ - 1);
  size_t j = idx >> 10;
  fTr[idx] = (bf16_t)ffr[e * E_ + j];
  fTi[idx] = (bf16_t)ffi[e * E_ + j];
  if (idx < (size_t)V_ * E_) {
    wrb[idx] = (bf16_t)rwr[idx];
    wib[idx] = (bf16_t)rwi[idx];
  }
}

// ---------------------------------------------------------------------------
// Kernel 3: causal flash attention, one wave per (b,h,16-query tile).
// Fragment loads are batched ahead of the WMMAs so they cluster into one
// VMEM clause; s0/s1 and o_r/o_i form independent WMMA dependency chains.
// ---------------------------------------------------------------------------
__global__ __launch_bounds__(128) void attn_kernel(
    const bf16_t* __restrict__ qr, const bf16_t* __restrict__ qi,
    const bf16_t* __restrict__ kr, const bf16_t* __restrict__ ki,
    const bf16_t* __restrict__ vtr, const bf16_t* __restrict__ vti,
    const float* __restrict__ sr, const float* __restrict__ si,
    bf16_t* __restrict__ xr, bf16_t* __restrict__ xi) {
  __shared__ __align__(16) bf16_t Plds[4][16 * 32];
  const int lane = threadIdx.x & 31;
  const int wave = threadIdx.x >> 5;
  const int col = lane & 15;
  const int rowBase = (lane & 16) ? 8 : 0;
  const int wid = blockIdx.x * 4 + wave;  // 0..4095
  const int qt = wid & 63;
  const int h = (wid >> 6) & 15;
  const int b = wid >> 10;
  bf16_t* Pw = &Plds[wave][0];

  const size_t qbase = (size_t)(b * S_ + qt * 16 + col) * E_ + h * HD_;
  v16bf qAr[2], qAi[2];
  qAr[0] = frag_a(qr + qbase, lane);
  qAr[1] = frag_a(qr + qbase + 32, lane);
  qAi[0] = frag_a(qi + qbase, lane);
  qAi[1] = frag_a(qi + qbase + 32, lane);

  const bf16_t* kbr = kr + (size_t)(b * S_) * E_ + h * HD_;
  const bf16_t* kbi = ki + (size_t)(b * S_) * E_ + h * HD_;
  const bf16_t* vbr = vtr + (size_t)((b * H_ + h) * HD_) * S_;
  const bf16_t* vbi = vti + (size_t)((b * H_ + h) * HD_) * S_;

  v8f o_r[4], o_i[4];
#pragma unroll
  for (int g = 0; g < 4; ++g) {
    o_r[g] = vzero8();
    o_i[g] = vzero8();
  }
  float m[8], lsum[8];
#pragma unroll
  for (int v = 0; v < 8; ++v) {
    m[v] = -3.0e38f;
    lsum[v] = 0.0f;
  }

  const int qhi = qt * 16 + 15;
  for (int k0 = 0; k0 <= qhi; k0 += 32) {
    const size_t kr0 = (size_t)(k0 + col) * E_;       // keys group 0
    const size_t kr1 = (size_t)(k0 + 16 + col) * E_;  // keys group 1
    v8f s0 = vzero8(), s1 = vzero8();
#pragma unroll
    for (int kk = 0; kk < 2; ++kk) {
      // batch 4 fragment loads, then interleave the two WMMA chains
      v16bf fr0 = frag_b(kbr + kr0 + kk * 32, lane);
      v16bf fr1 = frag_b(kbr + kr1 + kk * 32, lane);
      v16bf fi0 = frag_b(kbi + kr0 + kk * 32, lane);
      v16bf fi1 = frag_b(kbi + kr1 + kk * 32, lane);
      s0 = wmma_bf16(qAr[kk], fr0, s0);
      s1 = wmma_bf16(qAr[kk], fr1, s1);
      s0 = wmma_bf16(qAi[kk], fi0, s0);
      s1 = wmma_bf16(qAi[kk], fi1, s1);
    }
    // ---- causal mask + online softmax (8/sqrt(hd) == 1.0) ----
    float alpha[8], p0a[8], p1a[8];
#pragma unroll
    for (int v = 0; v < 8; ++v) {
      int q = qt * 16 + rowBase + v;
      float a0 = ((k0 + col) <= q) ? s0[v] : -3.0e38f;
      float a1 = ((k0 + 16 + col) <= q) ? s1[v] : -3.0e38f;
      float t = fmaxf(a0, a1);
#pragma unroll
      for (int off = 1; off < 16; off <<= 1) t = fmaxf(t, __shfl_xor(t, off));
      float mn = fmaxf(m[v], t);
      alpha[v] = __expf(m[v] - mn);
      float p0 = __expf(a0 - mn);
      float p1 = __expf(a1 - mn);
      float rs = p0 + p1;
#pragma unroll
      for (int off = 1; off < 16; off <<= 1) rs += __shfl_xor(rs, off);
      lsum[v] = lsum[v] * alpha[v] + rs;
      m[v] = mn;
      p0a[v] = p0;
      p1a[v] = p1;
    }
#pragma unroll
    for (int g = 0; g < 4; ++g)
#pragma unroll
      for (int v = 0; v < 8; ++v) {
        o_r[g][v] *= alpha[v];
        o_i[g][v] *= alpha[v];
      }
    // ---- P: C-layout -> A-layout via LDS (in-order per-wave LDS pipe) ----
#pragma unroll
    for (int v = 0; v < 8; ++v) {
      Pw[(rowBase + v) * 32 + col] = (bf16_t)p0a[v];
      Pw[(rowBase + v) * 32 + 16 + col] = (bf16_t)p1a[v];
    }
    asm volatile("s_wait_dscnt 0" ::: "memory");
    v16bf pA = frag_a(Pw + col * 32, lane);
    // ---- O += P @ V: batch loads, interleave 4 accumulator chains ----
#pragma unroll
    for (int g = 0; g < 4; g += 2) {
      v16bf w0 = frag_b(vbr + (size_t)(g * 16 + col) * S_ + k0, lane);
      v16bf w1 = frag_b(vbi + (size_t)(g * 16 + col) * S_ + k0, lane);
      v16bf w2 = frag_b(vbr + (size_t)((g + 1) * 16 + col) * S_ + k0, lane);
      v16bf w3 = frag_b(vbi + (size_t)((g + 1) * 16 + col) * S_ + k0, lane);
      o_r[g] = wmma_bf16(pA, w0, o_r[g]);
      o_i[g] = wmma_bf16(pA, w1, o_i[g]);
      o_r[g + 1] = wmma_bf16(pA, w2, o_r[g + 1]);
      o_i[g + 1] = wmma_bf16(pA, w3, o_i[g + 1]);
    }
  }
  float inv[8];
#pragma unroll
  for (int v = 0; v < 8; ++v) inv[v] = 1.0f / lsum[v];
#pragma unroll
  for (int g = 0; g < 4; ++g)
#pragma unroll
    for (int v = 0; v < 8; ++v) {
      int q = qt * 16 + rowBase + v;
      size_t idx = (size_t)(b * S_ + q) * E_ + h * HD_ + g * 16 + col;
      xr[idx] = (bf16_t)(o_r[g][v] * inv[v] + sr[idx]);
      xi[idx] = (bf16_t)(o_i[g][v] * inv[v] + si[idx]);
    }
}

// ---------------------------------------------------------------------------
// Kernel 4: complex FF GEMM with TDM-staged B tiles.
// Block = 4 waves = 64(M) x 64(N) tile. Per 32-K step the block needs the
// same 64x32 bf16 ff^T tile (r+i = 8KB): wave 0 DMAs it to LDS with
// tensor_load_to_lds (double-buffered, s_wait_tensorcnt pipelined), all
// waves consume it. yr = Xr@Fr - Xi@Fi ; yi = Xr@Fi + Xi@Fr.
// ---------------------------------------------------------------------------
__global__ __launch_bounds__(128) void ff_gemm_kernel(
    const bf16_t* __restrict__ xr, const bf16_t* __restrict__ xi,
    const bf16_t* __restrict__ fTr, const bf16_t* __restrict__ fTi,
    float* __restrict__ yr, float* __restrict__ yi) {
  __shared__ __align__(16) bf16_t Btile[2][2][64 * 32];  // [buf][r/i][row*32+k]
  const int lane = threadIdx.x & 31;
  const int wave = threadIdx.x >> 5;
  const int col = lane & 15;
  const int rowBase = (lane & 16) ? 8 : 0;
  const int nt = blockIdx.x & 15;   // 16 tiles of 64 cols
  const int mtb = blockIdx.x >> 4;  // 64 blocks of 64 rows
  const int mrow = mtb * 64 + wave * 16;
  const int n0 = nt * 64;

  const bf16_t* aR = xr + (size_t)(mrow + col) * E_;
  const bf16_t* aI = xi + (size_t)(mrow + col) * E_;

  v8f accRR[4], accII[4], accI[4];
#pragma unroll
  for (int g = 0; g < 4; ++g) {
    accRR[g] = vzero8();
    accII[g] = vzero8();
    accI[g] = vzero8();
  }
  const int nsteps = E_ / 32;
#if HAVE_TDM
  if (wave == 0) {
    tdm_load_tile((unsigned)(uintptr_t)&Btile[0][0][0],
                  fTr + (size_t)n0 * E_, 64, 32, E_);
    tdm_load_tile((unsigned)(uintptr_t)&Btile[0][1][0],
                  fTi + (size_t)n0 * E_, 64, 32, E_);
  }
#endif
  for (int ks = 0; ks < nsteps; ++ks) {
    const int k0 = ks * 32;
    const int cur = ks & 1;
#if HAVE_TDM
    if (wave == 0) {
      if (ks + 1 < nsteps) {
        tdm_load_tile((unsigned)(uintptr_t)&Btile[cur ^ 1][0][0],
                      fTr + (size_t)n0 * E_ + k0 + 32, 64, 32, E_);
        tdm_load_tile((unsigned)(uintptr_t)&Btile[cur ^ 1][1][0],
                      fTi + (size_t)n0 * E_ + k0 + 32, 64, 32, E_);
        __builtin_amdgcn_s_wait_tensorcnt(2);  // current tile complete
      } else {
        __builtin_amdgcn_s_wait_tensorcnt(0);
      }
    }
    __syncthreads();
#else
    {  // fallback: cooperative copy, 16B chunks (8KB per step / 128 thr)
      for (int c = threadIdx.x * 4; c < threadIdx.x * 4 + 4; ++c) {
        int comp = c >> 8;
        int row = (c & 255) >> 2;
        int part = (c & 3) * 8;
        const bf16_t* g = (comp ? fTi : fTr) + (size_t)(n0 + row) * E_ + k0 + part;
        *(v8bf*)&Btile[cur][comp][row * 32 + part] = *(const v8bf*)g;
      }
      __syncthreads();
    }
#endif
    v16bf ar = frag_a(aR + k0, lane);
    v16bf ai = frag_a(aI + k0, lane);
    const bf16_t* btr = &Btile[cur][0][0];
    const bf16_t* bti = &Btile[cur][1][0];
    v16bf bfr[4], bfi[4];
#pragma unroll
    for (int g = 0; g < 4; ++g) {
      bfr[g] = frag_b(btr + (g * 16 + col) * 32, lane);
      bfi[g] = frag_b(bti + (g * 16 + col) * 32, lane);
    }
#pragma unroll
    for (int g = 0; g < 4; ++g) {
      accRR[g] = wmma_bf16(ar, bfr[g], accRR[g]);
      accII[g] = wmma_bf16(ai, bfi[g], accII[g]);
      accI[g] = wmma_bf16(ar, bfi[g], accI[g]);
      accI[g] = wmma_bf16(ai, bfr[g], accI[g]);
    }
    __syncthreads();
  }
#pragma unroll
  for (int g = 0; g < 4; ++g)
#pragma unroll
    for (int v = 0; v < 8; ++v) {
      size_t row = (size_t)(mrow + rowBase + v);
      int cc = n0 + g * 16 + col;
      yr[row * E_ + cc] = accRR[g][v] - accII[g][v];
      yi[row * E_ + cc] = accI[g][v];
    }
}

// ---------------------------------------------------------------------------
// Kernel 5: complex_norm per row (mean/std of |z| with ddof=1, tanh scaling).
// ---------------------------------------------------------------------------
__global__ __launch_bounds__(256) void norm_kernel(
    const float* __restrict__ yr, const float* __restrict__ yi,
    bf16_t* __restrict__ nr, bf16_t* __restrict__ ni) {
  __shared__ float sh1[256], sh2[256];
  const int row = blockIdx.x;
  const int t = threadIdx.x;
  const float* pr = yr + (size_t)row * E_;
  const float* pi = yi + (size_t)row * E_;
  float rv[4], iv[4], mg[4];
  float s = 0.0f, ss = 0.0f;
#pragma unroll
  for (int j = 0; j < 4; ++j) {
    int c = t + j * 256;
    rv[j] = pr[c];
    iv[j] = pi[c];
    float mag = sqrtf(rv[j] * rv[j] + iv[j] * iv[j]);
    mg[j] = mag;
    s += mag;
    ss += mag * mag;
  }
  sh1[t] = s;
  sh2[t] = ss;
  __syncthreads();
  for (int off = 128; off > 0; off >>= 1) {
    if (t < off) {
      sh1[t] += sh1[t + off];
      sh2[t] += sh2[t + off];
    }
    __syncthreads();
  }
  float mean = sh1[0] * (1.0f / E_);
  float var = (sh2[0] - (float)E_ * mean * mean) * (1.0f / (E_ - 1));
  float stdv = sqrtf(fmaxf(var, 0.0f));
#pragma unroll
  for (int j = 0; j < 4; ++j) {
    int c = t + j * 256;
    float nm = (mg[j] - mean) / (stdv + EPS_);
    float sc = tanhf(nm) / (mg[j] + EPS_);
    nr[(size_t)row * E_ + c] = (bf16_t)(rv[j] * sc);
    ni[(size_t)row * E_ + c] = (bf16_t)(iv[j] * sc);
  }
}

// ---------------------------------------------------------------------------
// Kernel 6: readout GEMM. out = nR @ Wr^T + nI @ Wi^T + (br + bi).
// ---------------------------------------------------------------------------
__global__ __launch_bounds__(128) void ro_gemm_kernel(
    const bf16_t* __restrict__ nr, const bf16_t* __restrict__ ni,
    const bf16_t* __restrict__ wr, const bf16_t* __restrict__ wi,
    const float* __restrict__ br, const float* __restrict__ bi,
    float* __restrict__ out) {
  const int lane = threadIdx.x & 31;
  const int col = lane & 15;
  const int rowBase = (lane & 16) ? 8 : 0;
  const int wid = blockIdx.x * 4 + (threadIdx.x >> 5);  // 0..1023
  const int nt = wid & 3;                               // 4 tiles of 64 vocab
  const int mt = wid >> 2;                              // 256 tiles of 16 rows
  const bf16_t* aR = nr + (size_t)(mt * 16 + col) * E_;
  const bf16_t* aI = ni + (size_t)(mt * 16 + col) * E_;
  v8f acc[4];
#pragma unroll
  for (int g = 0; g < 4; ++g) acc[g] = vzero8();
  for (int k0 = 0; k0 < E_; k0 += 32) {
    v16bf ar = frag_a(aR + k0, lane);
    v16bf ai = frag_a(aI + k0, lane);
    v16bf wbr[4], wbi[4];
#pragma unroll
    for (int g = 0; g < 4; ++g) {
      int n = nt * 64 + g * 16 + col;
      wbr[g] = frag_b(wr + (size_t)n * E_ + k0, lane);
      wbi[g] = frag_b(wi + (size_t)n * E_ + k0, lane);
    }
#pragma unroll
    for (int g = 0; g < 4; ++g) {
      acc[g] = wmma_bf16(ar, wbr[g], acc[g]);
      acc[g] = wmma_bf16(ai, wbi[g], acc[g]);
    }
  }
#pragma unroll
  for (int g = 0; g < 4; ++g)
#pragma unroll
    for (int v = 0; v < 8; ++v) {
      size_t row = (size_t)(mt * 16 + rowBase + v);
      int cc = nt * 64 + g * 16 + col;
      out[row * V_ + cc] = acc[g][v] + br[cc] + bi[cc];
    }
}

// ---------------------------------------------------------------------------
extern "C" void kernel_launch(void* const* d_in, const int* in_sizes, int n_in,
                              void* d_out, int out_size, void* d_ws,
                              size_t ws_size, hipStream_t stream) {
  const int* x = (const int*)d_in[0];
  const float* emb = (const float*)d_in[1];
  const float* posp = (const float*)d_in[2];
  const float* qrot = (const float*)d_in[3];
  const float* krot = (const float*)d_in[4];
  const float* vrot = (const float*)d_in[5];
  const float* ffr = (const float*)d_in[6];
  const float* ffi = (const float*)d_in[7];
  const float* rwr = (const float*)d_in[8];
  const float* rbr = (const float*)d_in[9];
  const float* rwi = (const float*)d_in[10];
  const float* rbi = (const float*)d_in[11];
  float* out = (float*)d_out;

  const size_t NE = (size_t)B_ * S_ * E_;  // 4Mi elements
  char* ws = (char*)d_ws;
  size_t off = 0;
  auto take = [&](size_t bytes) {
    void* p = (void*)(ws + off);
    off += (bytes + 255) & ~(size_t)255;
    return p;
  };
  float* sr = (float*)take(NE * 4);
  float* si = (float*)take(NE * 4);
  bf16_t* qr = (bf16_t*)take(NE * 2);
  bf16_t* qi = (bf16_t*)take(NE * 2);
  bf16_t* kr = (bf16_t*)take(NE * 2);
  bf16_t* ki = (bf16_t*)take(NE * 2);
  bf16_t* vtr = (bf16_t*)take(NE * 2);
  bf16_t* vti = (bf16_t*)take(NE * 2);
  bf16_t* xrb = (bf16_t*)take(NE * 2);
  bf16_t* xib = (bf16_t*)take(NE * 2);
  bf16_t* fTr = (bf16_t*)take((size_t)E_ * E_ * 2);
  bf16_t* fTi = (bf16_t*)take((size_t)E_ * E_ * 2);
  float* yr = (float*)take(NE * 4);
  float* yi = (float*)take(NE * 4);
  bf16_t* nrm_r = (bf16_t*)take(NE * 2);
  bf16_t* nrm_i = (bf16_t*)take(NE * 2);
  bf16_t* wrb = (bf16_t*)take((size_t)V_ * E_ * 2);
  bf16_t* wib = (bf16_t*)take((size_t)V_ * E_ * 2);

  embed_kernel<<<(int)(NE / 256), 256, 0, stream>>>(
      x, emb, posp, qrot, krot, vrot, sr, si, qr, qi, kr, ki, vtr, vti);
  prep_kernel<<<(E_ * E_) / 256, 256, 0, stream>>>(ffr, ffi, rwr, rwi, fTr,
                                                   fTi, wrb, wib);
  attn_kernel<<<(B_ * H_ * (S_ / 16)) / 4, 128, 0, stream>>>(
      qr, qi, kr, ki, vtr, vti, sr, si, xrb, xib);
  ff_gemm_kernel<<<(E_ / 64) * (B_ * S_ / 64), 128, 0, stream>>>(
      xrb, xib, fTr, fTi, yr, yi);
  norm_kernel<<<B_ * S_, 256, 0, stream>>>(yr, yi, nrm_r, nrm_i);
  ro_gemm_kernel<<<((B_ * S_ / 16) * (V_ / 64)) / 4, 128, 0, stream>>>(
      nrm_r, nrm_i, wrb, wib, rbr, rbi, out);
}
